// PointNet2Regressor_SoPs_15324443312192
// MI455X (gfx1250) — matrix operand order repaired and checked by reading, not compiled, not run on hardware
//
#include <hip/hip_runtime.h>
#include <hip/hip_bf16.h>
#include <math.h>

typedef __attribute__((ext_vector_type(16))) _Float16 v16h;
typedef __attribute__((ext_vector_type(8)))  _Float16 v8h;
typedef __attribute__((ext_vector_type(8)))  float    v8f;

#define BN_INV 0.9999950000374997f   // 1/sqrt(1 + 1e-5)

// ---------------------------------------------------------------------------
// WMMA helpers (gfx1250 wave32 layouts, CDNA5 ISA 7.12.2)
// ---------------------------------------------------------------------------
__device__ __forceinline__ v8f wmma16(v16h a, v16h b, v8f c) {
  return __builtin_amdgcn_wmma_f32_16x16x32_f16(false, a, false, b, (short)0, c,
                                                false, false);
}

__device__ __forceinline__ v16h make16(v8h lo, v8h hi) {
  return __builtin_shufflevector(lo, hi, 0, 1, 2, 3, 4, 5, 6, 7,
                                 8, 9, 10, 11, 12, 13, 14, 15);
}

// A fragment (16x32 f16) from LDS, row-major, stride multiple of 8 halfs so
// every fragment pointer is 16B aligned -> two ds_load_b128.
// Lane L (L<16): row=L, elems 0..15 -> K=k0..k0+15
// Lane L+16   : row=L, elems 0..15 -> K=k0+16..k0+31
__device__ __forceinline__ v16h load_a_lds(const _Float16* X, int stride,
                                           int m0, int k0, int lane) {
  const _Float16* p = X + (m0 + (lane & 15)) * stride + k0 + ((lane >> 4) << 4);
  const v8h* q = (const v8h*)p;
  return make16(q[0], q[1]);
}

// A fragment from global f32 (row-major, stride K).  Row index is CLAMPED
// (out-of-range rows feed only masked-out outputs), so no load predication.
__device__ __forceinline__ v16h load_a_g(const float* A, int M, int K,
                                         int m0, int k0, int lane, bool full) {
  int row = m0 + (lane & 15);
  row = row < M ? row : M - 1;
  int kb = k0 + ((lane >> 4) << 4);
  const float* p = A + (size_t)row * K + kb;
  v16h a;
  if (full) {            // k0+32 <= K : unguarded contiguous loads
#pragma unroll
    for (int e = 0; e < 8; ++e) {
      a[e]     = (_Float16)p[e];
      a[e + 8] = (_Float16)p[e + 8];
    }
  } else {               // tail: clamp K index, select value (v_cndmask)
    const float* base = A + (size_t)row * K;
#pragma unroll
    for (int e = 0; e < 8; ++e) {
      int k1 = kb + e, k2 = kb + e + 8;
      float v1 = base[k1 < K ? k1 : K - 1];
      float v2 = base[k2 < K ? k2 : K - 1];
      a[e]     = (_Float16)(k1 < K ? v1 : 0.f);
      a[e + 8] = (_Float16)(k2 < K ? v2 : 0.f);
    }
  }
  return a;
}

// B fragment (32x16): Bmat[k][n] = W[n][k], W is [Ncols][Cin] row-major f32.
// Column CLAMPED; K guarded only in the tail via value-select.
__device__ __forceinline__ v16h load_b_g(const float* W, int Cin, int Ncols,
                                         int n0, int k0, int lane, bool full) {
  int col = n0 + (lane & 15);
  col = col < Ncols ? col : Ncols - 1;
  int kb = k0 + ((lane >> 4) << 4);
  const float* p = W + (size_t)col * Cin + kb;
  v16h b;
  if (full) {
#pragma unroll
    for (int e = 0; e < 16; ++e) b[e] = (_Float16)p[e];
  } else {
    const float* base = W + (size_t)col * Cin;
#pragma unroll
    for (int e = 0; e < 16; ++e) {
      int kk = kb + e;
      float v = base[kk < Cin ? kk : Cin - 1];
      b[e] = (_Float16)(kk < Cin ? v : 0.f);
    }
  }
  return b;
}

// Compile-time-guarded B load for the fused kernels (Cin constexpr, k0 becomes
// a constant after full unrolling, cols always in range).
template <int Cin>
__device__ __forceinline__ v16h load_b_wc(const float* W, int n0, int k0,
                                          int lane) {
  int col = n0 + (lane & 15);
  int kb = k0 + ((lane >> 4) << 4);
  const float* p = W + (size_t)col * Cin + kb;
  v16h b;
  if (k0 + 32 <= Cin) {
#pragma unroll
    for (int e = 0; e < 16; ++e) b[e] = (_Float16)p[e];
  } else {
    const float* base = W + (size_t)col * Cin;
#pragma unroll
    for (int e = 0; e < 16; ++e) {
      int kk = kb + e;
      float v = base[kk < Cin ? kk : Cin - 1];
      b[e] = (_Float16)(kk < Cin ? v : 0.f);
    }
  }
  return b;
}

// ---------------------------------------------------------------------------
// Kernel: transpose xyz (B,3,N) -> pts (B,N,3)
// ---------------------------------------------------------------------------
__global__ __launch_bounds__(256) void transpose_xyz_kernel(
    const float* __restrict__ xyz, float* __restrict__ pts, int Bb, int N) {
  int i = blockIdx.x * blockDim.x + threadIdx.x;
  int tot = Bb * N * 3;
  if (i >= tot) return;
  int b = i / (N * 3);
  int r = i - b * (N * 3);
  int n = r / 3;
  int c = r - n * 3;
  pts[i] = xyz[((size_t)b * 3 + c) * N + n];
}

// ---------------------------------------------------------------------------
// Kernel: farthest point sampling. One block (256 thr) per batch.
// scan semantics: output[it] = incoming `far` (first = 0); argmax ties -> lowest idx.
// ---------------------------------------------------------------------------
template <int PPT>
__global__ __launch_bounds__(256) void fps_kernel(
    const float* __restrict__ pts, int N, int npoint,
    int* __restrict__ fidx, float* __restrict__ cents) {
  const int b = blockIdx.x;
  const int tid = threadIdx.x;
  const float* P = pts + (size_t)b * N * 3;
  float dist[PPT];
#pragma unroll
  for (int i = 0; i < PPT; ++i) dist[i] = 1e10f;

  __shared__ float sv[256];
  __shared__ int   si[256];
  __shared__ float cen[3];

  int far = 0;
  for (int it = 0; it < npoint; ++it) {
    if (tid == 0) {
      fidx[b * npoint + it] = far;
      float x = P[far * 3 + 0], y = P[far * 3 + 1], z = P[far * 3 + 2];
      cents[((size_t)b * npoint + it) * 3 + 0] = x;
      cents[((size_t)b * npoint + it) * 3 + 1] = y;
      cents[((size_t)b * npoint + it) * 3 + 2] = z;
      cen[0] = x; cen[1] = y; cen[2] = z;
    }
    __syncthreads();
    float cx = cen[0], cy = cen[1], cz = cen[2];
    float bestv = -1.f;
    int   besti = N;
#pragma unroll
    for (int i = 0; i < PPT; ++i) {
      int j = tid + i * 256;
      if (j < N) {
        float dx = P[j * 3 + 0] - cx;
        float dy = P[j * 3 + 1] - cy;
        float dz = P[j * 3 + 2] - cz;
        float d = dx * dx + dy * dy + dz * dz;
        float nd = fminf(dist[i], d);
        dist[i] = nd;
        if (nd > bestv) { bestv = nd; besti = j; }  // j increasing -> first max
      }
    }
    sv[tid] = bestv; si[tid] = besti;
    __syncthreads();
    for (int off = 128; off > 0; off >>= 1) {
      if (tid < off) {
        float v2 = sv[tid + off]; int i2 = si[tid + off];
        if (v2 > sv[tid] || (v2 == sv[tid] && i2 < si[tid])) {
          sv[tid] = v2; si[tid] = i2;
        }
      }
      __syncthreads();
    }
    far = si[0];
    // next-iteration barrier orders this read vs. sv/si rewrites
  }
}

// ---------------------------------------------------------------------------
// Kernel: ball query. One wave32 per centroid; ballot-prefix compaction keeps
// the first `nsample` in-radius indices ascending (reference semantics),
// padding with the first hit.
// ---------------------------------------------------------------------------
__global__ __launch_bounds__(256) void ball_query_kernel(
    const float* __restrict__ pts, int N,
    const float* __restrict__ cents, int S, int Bb,
    float r2, int nsample, int* __restrict__ gidx) {
  int wid  = (blockIdx.x * blockDim.x + threadIdx.x) >> 5;
  int lane = threadIdx.x & 31;
  if (wid >= Bb * S) return;
  int b = wid / S;
  const float* P = pts + (size_t)b * N * 3;
  float cx = cents[(size_t)wid * 3 + 0];
  float cy = cents[(size_t)wid * 3 + 1];
  float cz = cents[(size_t)wid * 3 + 2];
  int* out = gidx + (size_t)wid * nsample;

  int cnt = 0;
  for (int base = 0; base < N && cnt < nsample; base += 32) {
    int j = base + lane;
    bool in = false;
    if (j < N) {
      float dx = P[j * 3 + 0] - cx;
      float dy = P[j * 3 + 1] - cy;
      float dz = P[j * 3 + 2] - cz;
      in = (dx * dx + dy * dy + dz * dz) <= r2;
    }
    unsigned m = (unsigned)__ballot(in);
    int pos = cnt + __popc(m & ((1u << lane) - 1u));
    if (in && pos < nsample) out[pos] = j;
    cnt += __popc(m);
  }
  if (lane == 0 && cnt < nsample) {
    int first = out[0];   // centroid itself always in radius -> cnt >= 1
    for (int p = cnt; p < nsample; ++p) out[p] = first;
  }
}

// ---------------------------------------------------------------------------
// Device: one MLP layer, X(LDS f16, NSxKP) * W^T (global f32 [Cout][Cin])
// -> Y(LDS f16) with fused bias + BN + ReLU. 8 waves per block.
// ---------------------------------------------------------------------------
template <int NS, int KP, int Cin, int Cout>
__device__ __forceinline__ void mlp_layer_lds(
    const _Float16* X, const float* W, const float* bias,
    const float* gm, const float* bt, _Float16* Y) {
  const int lane = threadIdx.x & 31;
  const int wave = threadIdx.x >> 5;
  constexpr int TM = NS / 16, TN = Cout / 16;
  for (int t = wave; t < TM * TN; t += 8) {
    int tm = t / TN, tn = t - tm * TN;
    v8f acc = {0.f, 0.f, 0.f, 0.f, 0.f, 0.f, 0.f, 0.f};
#pragma unroll
    for (int k0 = 0; k0 < KP; k0 += 32) {
      v16h av = load_a_lds(X, KP, tm * 16, k0, lane);
      v16h bv = load_b_wc<Cin>(W, tn * 16, k0, lane);
      acc = wmma16(av, bv, acc);
    }
    int n  = tn * 16 + (lane & 15);
    int mb = tm * 16 + ((lane >> 4) << 3);
    float sg = gm[n] * BN_INV, sb = bias[n], st = bt[n];
#pragma unroll
    for (int r = 0; r < 8; ++r) {
      float y = sg * (acc[r] + sb) + st;
      Y[(mb + r) * Cout + n] = (_Float16)(y > 0.f ? y : 0.f);
    }
  }
}

// Final layer + max-pool over the NS rows, entirely in registers.
template <int NS, int Cin, int Cout>
__device__ __forceinline__ void mlp_final_max(
    const _Float16* X, const float* W, const float* bias,
    const float* gm, const float* bt, float* outrow) {
  const int lane = threadIdx.x & 31;
  const int wave = threadIdx.x >> 5;
  constexpr int TM = NS / 16, TN = Cout / 16;
  for (int tn = wave; tn < TN; tn += 8) {
    int n = tn * 16 + (lane & 15);
    float sg = gm[n] * BN_INV, sb = bias[n], st = bt[n];
    float cm = -1e30f;
    for (int tm = 0; tm < TM; ++tm) {
      v8f acc = {0.f, 0.f, 0.f, 0.f, 0.f, 0.f, 0.f, 0.f};
#pragma unroll
      for (int k0 = 0; k0 < Cin; k0 += 32) {
        v16h av = load_a_lds(X, Cin, tm * 16, k0, lane);
        v16h bv = load_b_wc<Cin>(W, tn * 16, k0, lane);
        acc = wmma16(av, bv, acc);
      }
#pragma unroll
      for (int r = 0; r < 8; ++r) {
        float y = sg * (acc[r] + sb) + st;
        cm = fmaxf(cm, y > 0.f ? y : 0.f);
      }
    }
    cm = fmaxf(cm, __shfl_xor(cm, 16));  // merge M=r and M=r+8 lane halves
    if (lane < 16) outrow[n] = cm;
  }
}

// ---------------------------------------------------------------------------
// Fused set-abstraction: one block per (batch, centroid). Gather grouped input
// (rel coords + optional features) into LDS f16, 3-layer WMMA MLP, max-pool.
// ---------------------------------------------------------------------------
template <int NS, int C0, int C0P, int C1, int C2, int C3>
__global__ __launch_bounds__(256) void sa_fused_kernel(
    const float* __restrict__ pts, int Npts,
    const float* __restrict__ feat, int Cfeat,
    const float* __restrict__ cents, int S,
    const int* __restrict__ gidx,
    const float* __restrict__ W1, const float* __restrict__ b1,
    const float* __restrict__ g1, const float* __restrict__ t1,
    const float* __restrict__ W2, const float* __restrict__ b2,
    const float* __restrict__ g2, const float* __restrict__ t2,
    const float* __restrict__ W3, const float* __restrict__ b3,
    const float* __restrict__ g3, const float* __restrict__ t3,
    float* __restrict__ outfeat) {
  __shared__ __align__(16) _Float16 sX[NS * C0P];
  __shared__ __align__(16) _Float16 sH1[NS * C1];
  __shared__ __align__(16) _Float16 sH2[NS * C2];

  const int grp = blockIdx.x;
  const int b = grp / S;
  const int tid = threadIdx.x;

  const float cx = cents[(size_t)grp * 3 + 0];
  const float cy = cents[(size_t)grp * 3 + 1];
  const float cz = cents[(size_t)grp * 3 + 2];
  const int* gr = gidx + (size_t)grp * NS;
  const float* Pb = pts + (size_t)b * Npts * 3;

  for (int i = tid; i < NS * C0P; i += 256) {
    int r = i / C0P;
    int c = i - r * C0P;
    float v = 0.f;
    int j = gr[r];
    if (c < 3) {
      v = Pb[j * 3 + c] - (c == 0 ? cx : (c == 1 ? cy : cz));
    } else if (c < C0) {
      v = feat[((size_t)b * Npts + j) * Cfeat + (c - 3)];
    }
    sX[i] = (_Float16)v;
  }
  __syncthreads();
  mlp_layer_lds<NS, C0P, C0, C1>(sX, W1, b1, g1, t1, sH1);
  __syncthreads();
  mlp_layer_lds<NS, C1, C1, C2>(sH1, W2, b2, g2, t2, sH2);
  __syncthreads();
  mlp_final_max<NS, C2, C3>(sH2, W3, b3, g3, t3, outfeat + (size_t)grp * C3);
}

// ---------------------------------------------------------------------------
// Generic WMMA GEMM: Y[M,N] = epilogue(A[M,K] * W[N,K]^T).
// One wave per 16x64 tile (4 accumulators, A fragment reused 4x).
// mode 0: Y = acc + bias ; mode 1: Y = relu(gm*BN_INV*(acc+bias)+bt)
// ---------------------------------------------------------------------------
__global__ __launch_bounds__(256) void gemm_bn_kernel(
    const float* __restrict__ A, int M, int K,
    const float* __restrict__ W, const float* __restrict__ bias,
    const float* __restrict__ gm, const float* __restrict__ bt,
    int N, float* __restrict__ Y, int mode) {
  int wid  = (blockIdx.x * blockDim.x + threadIdx.x) >> 5;
  int lane = threadIdx.x & 31;
  int ntm = (M + 15) >> 4;
  int ntn = (N + 63) >> 6;
  if (wid >= ntm * ntn) return;   // wave-uniform: EXEC stays full for WMMA
  int tm = wid / ntn;
  int n0 = (wid - tm * ntn) * 64;
  int m0 = tm * 16;

  v8f acc[4];
#pragma unroll
  for (int u = 0; u < 4; ++u)
    acc[u] = (v8f){0.f, 0.f, 0.f, 0.f, 0.f, 0.f, 0.f, 0.f};

  int kmain = K & ~31;
  for (int k0 = 0; k0 < kmain; k0 += 32) {
    v16h av = load_a_g(A, M, K, m0, k0, lane, true);
#pragma unroll
    for (int u = 0; u < 4; ++u) {
      v16h bv = load_b_g(W, K, N, n0 + u * 16, k0, lane, true);
      acc[u] = wmma16(av, bv, acc[u]);
    }
  }
  if (kmain < K) {
    v16h av = load_a_g(A, M, K, m0, kmain, lane, false);
#pragma unroll
    for (int u = 0; u < 4; ++u) {
      v16h bv = load_b_g(W, K, N, n0 + u * 16, kmain, lane, false);
      acc[u] = wmma16(av, bv, acc[u]);
    }
  }

  int lane15 = lane & 15;
  int mb = m0 + ((lane >> 4) << 3);
#pragma unroll
  for (int u = 0; u < 4; ++u) {
    int n = n0 + u * 16 + lane15;
    if (n < N) {
      float sb = bias[n];
      float sg = 0.f, st = 0.f;
      if (mode) { sg = gm[n] * BN_INV; st = bt[n]; }
#pragma unroll
      for (int r = 0; r < 8; ++r) {
        int m = mb + r;
        if (m < M) {
          float y = acc[u][r] + sb;
          if (mode) { y = sg * y + st; y = y > 0.f ? y : 0.f; }
          Y[(size_t)m * N + n] = y;
        }
      }
    }
  }
}

// ---------------------------------------------------------------------------
// SA3 input: concat(l2_xyz (rows,3), l2_feat (rows,C)) -> (rows, 3+C)
// ---------------------------------------------------------------------------
__global__ __launch_bounds__(256) void concat3_kernel(
    const float* __restrict__ cents, const float* __restrict__ feat,
    int rows, int C, float* __restrict__ out) {
  int i = blockIdx.x * blockDim.x + threadIdx.x;
  int W = 3 + C;
  if (i >= rows * W) return;
  int r = i / W, c = i - r * W;
  out[i] = (c < 3) ? cents[(size_t)r * 3 + c] : feat[(size_t)r * C + (c - 3)];
}

// Max over `rows` rows within each of `groups` groups.
__global__ __launch_bounds__(256) void maxpool_rows_kernel(
    const float* __restrict__ X, int groups, int rows, int C,
    float* __restrict__ Y) {
  int i = blockIdx.x * blockDim.x + threadIdx.x;
  if (i >= groups * C) return;
  int g = i / C, c = i - g * C;
  const float* p = X + ((size_t)g * rows) * C + c;
  float m = -1e30f;
  for (int r = 0; r < rows; ++r) m = fmaxf(m, p[(size_t)r * C]);
  Y[i] = m;
}

// Head: trans passthrough, normals = tanh -> L2-normalize; concat (B,10,6).
__global__ __launch_bounds__(256) void finalize_kernel(
    const float* __restrict__ trans, const float* __restrict__ nvec,
    float* __restrict__ out, int nv) {
  int i = blockIdx.x * blockDim.x + threadIdx.x;
  if (i >= nv) return;
  float tx = trans[i * 3 + 0], ty = trans[i * 3 + 1], tz = trans[i * 3 + 2];
  float nx = tanhf(nvec[i * 3 + 0]);
  float ny = tanhf(nvec[i * 3 + 1]);
  float nz = tanhf(nvec[i * 3 + 2]);
  float nrm = fmaxf(sqrtf(nx * nx + ny * ny + nz * nz), 1e-12f);
  float s = 1.0f / nrm;   // WEIGHT_ORIENT = 1.0
  out[i * 6 + 0] = tx; out[i * 6 + 1] = ty; out[i * 6 + 2] = tz;
  out[i * 6 + 3] = nx * s; out[i * 6 + 4] = ny * s; out[i * 6 + 5] = nz * s;
}

// ---------------------------------------------------------------------------
// Host orchestration
// ---------------------------------------------------------------------------
static inline void launch_gemm(const float* A, int M, int K, const float* W,
                               const float* bias, const float* gm, const float* bt,
                               int N, float* Y, int mode, hipStream_t s) {
  int tiles = ((M + 15) / 16) * ((N + 63) / 64);
  int blocks = (tiles * 32 + 255) / 256;
  gemm_bn_kernel<<<blocks, 256, 0, s>>>(A, M, K, W, bias, gm, bt, N, Y, mode);
}

extern "C" void kernel_launch(void* const* d_in, const int* in_sizes, int n_in,
                              void* d_out, int out_size, void* d_ws, size_t ws_size,
                              hipStream_t stream) {
  (void)in_sizes; (void)n_in; (void)out_size; (void)ws_size;
  const int Bb = 16, N = 16384, S1 = 512, NS1 = 32, S2 = 128, NS2 = 64;

  const float* xyz = (const float*)d_in[0];
  const float* s1W1 = (const float*)d_in[1];  const float* s1b1 = (const float*)d_in[2];
  const float* s1g1 = (const float*)d_in[3];  const float* s1t1 = (const float*)d_in[4];
  const float* s1W2 = (const float*)d_in[5];  const float* s1b2 = (const float*)d_in[6];
  const float* s1g2 = (const float*)d_in[7];  const float* s1t2 = (const float*)d_in[8];
  const float* s1W3 = (const float*)d_in[9];  const float* s1b3 = (const float*)d_in[10];
  const float* s1g3 = (const float*)d_in[11]; const float* s1t3 = (const float*)d_in[12];
  const float* s2W1 = (const float*)d_in[13]; const float* s2b1 = (const float*)d_in[14];
  const float* s2g1 = (const float*)d_in[15]; const float* s2t1 = (const float*)d_in[16];
  const float* s2W2 = (const float*)d_in[17]; const float* s2b2 = (const float*)d_in[18];
  const float* s2g2 = (const float*)d_in[19]; const float* s2t2 = (const float*)d_in[20];
  const float* s2W3 = (const float*)d_in[21]; const float* s2b3 = (const float*)d_in[22];
  const float* s2g3 = (const float*)d_in[23]; const float* s2t3 = (const float*)d_in[24];
  const float* s3W1 = (const float*)d_in[25]; const float* s3b1 = (const float*)d_in[26];
  const float* s3g1 = (const float*)d_in[27]; const float* s3t1 = (const float*)d_in[28];
  const float* s3W2 = (const float*)d_in[29]; const float* s3b2 = (const float*)d_in[30];
  const float* s3g2 = (const float*)d_in[31]; const float* s3t2 = (const float*)d_in[32];
  const float* s3W3 = (const float*)d_in[33]; const float* s3b3 = (const float*)d_in[34];
  const float* s3g3 = (const float*)d_in[35]; const float* s3t3 = (const float*)d_in[36];
  const float* fc1_w = (const float*)d_in[37]; const float* fc1_b = (const float*)d_in[38];
  const float* bn1_g = (const float*)d_in[39]; const float* bn1_b = (const float*)d_in[40];
  const float* fc2_w = (const float*)d_in[41]; const float* fc2_b = (const float*)d_in[42];
  const float* bn2_g = (const float*)d_in[43]; const float* bn2_b = (const float*)d_in[44];
  const float* fc3_w = (const float*)d_in[45]; const float* fc3_b = (const float*)d_in[46];
  const float* fcn_w = (const float*)d_in[47]; const float* fcn_b = (const float*)d_in[48];

  char* wsp = (char*)d_ws;
  auto alloc = [&](size_t bytes) -> void* {
    void* p = (void*)wsp;
    wsp += (bytes + 255) & ~(size_t)255;
    return p;
  };
  float* pts1   = (float*)alloc((size_t)Bb * N * 3 * 4);
  int*   fidx1  = (int*)  alloc((size_t)Bb * S1 * 4);
  float* cent1  = (float*)alloc((size_t)Bb * S1 * 3 * 4);
  int*   gidx1  = (int*)  alloc((size_t)Bb * S1 * NS1 * 4);
  float* l1feat = (float*)alloc((size_t)Bb * S1 * 128 * 4);
  int*   fidx2  = (int*)  alloc((size_t)Bb * S2 * 4);
  float* cent2  = (float*)alloc((size_t)Bb * S2 * 3 * 4);
  int*   gidx2  = (int*)  alloc((size_t)Bb * S2 * NS2 * 4);
  float* l2feat = (float*)alloc((size_t)Bb * S2 * 256 * 4);
  float* gin3   = (float*)alloc((size_t)Bb * S2 * 259 * 4);
  float* h3a    = (float*)alloc((size_t)Bb * S2 * 256 * 4);
  float* h3b    = (float*)alloc((size_t)Bb * S2 * 512 * 4);
  float* h3c    = (float*)alloc((size_t)Bb * S2 * 1024 * 4);
  float* gfeat  = (float*)alloc((size_t)Bb * 1024 * 4);
  float* fc1o   = (float*)alloc((size_t)Bb * 1024 * 4);
  float* fc2o   = (float*)alloc((size_t)Bb * 1024 * 4);
  float* transo = (float*)alloc((size_t)Bb * 30 * 4);
  float* nveco  = (float*)alloc((size_t)Bb * 30 * 4);

  {
    int tot = Bb * N * 3;
    transpose_xyz_kernel<<<(tot + 255) / 256, 256, 0, stream>>>(xyz, pts1, Bb, N);
  }
  fps_kernel<64><<<Bb, 256, 0, stream>>>(pts1, N, S1, fidx1, cent1);
  {
    int waves = Bb * S1;
    ball_query_kernel<<<(waves * 32 + 255) / 256, 256, 0, stream>>>(
        pts1, N, cent1, S1, Bb, 0.04f, NS1, gidx1);
  }
  sa_fused_kernel<NS1, 3, 32, 64, 64, 128><<<Bb * S1, 256, 0, stream>>>(
      pts1, N, (const float*)nullptr, 0, cent1, S1, gidx1,
      s1W1, s1b1, s1g1, s1t1, s1W2, s1b2, s1g2, s1t2, s1W3, s1b3, s1g3, s1t3,
      l1feat);
  fps_kernel<2><<<Bb, 256, 0, stream>>>(cent1, S1, S2, fidx2, cent2);
  {
    int waves = Bb * S2;
    ball_query_kernel<<<(waves * 32 + 255) / 256, 256, 0, stream>>>(
        cent1, S1, cent2, S2, Bb, 0.16f, NS2, gidx2);
  }
  sa_fused_kernel<NS2, 131, 160, 128, 128, 256><<<Bb * S2, 256, 0, stream>>>(
      cent1, S1, l1feat, 128, cent2, S2, gidx2,
      s2W1, s2b1, s2g1, s2t1, s2W2, s2b2, s2g2, s2t2, s2W3, s2b3, s2g3, s2t3,
      l2feat);
  {
    int rows = Bb * S2, tot = rows * 259;
    concat3_kernel<<<(tot + 255) / 256, 256, 0, stream>>>(cent2, l2feat, rows, 256, gin3);
    launch_gemm(gin3, rows, 259, s3W1, s3b1, s3g1, s3t1, 256, h3a, 1, stream);
    launch_gemm(h3a, rows, 256, s3W2, s3b2, s3g2, s3t2, 512, h3b, 1, stream);
    launch_gemm(h3b, rows, 512, s3W3, s3b3, s3g3, s3t3, 1024, h3c, 1, stream);
    int tp = Bb * 1024;
    maxpool_rows_kernel<<<(tp + 255) / 256, 256, 0, stream>>>(h3c, Bb, S2, 1024, gfeat);
  }
  launch_gemm(gfeat, Bb, 1024, fc1_w, fc1_b, bn1_g, bn1_b, 1024, fc1o, 1, stream);
  launch_gemm(fc1o,  Bb, 1024, fc2_w, fc2_b, bn2_g, bn2_b, 1024, fc2o, 1, stream);
  launch_gemm(fc2o,  Bb, 1024, fc3_w, fc3_b, nullptr, nullptr, 30, transo, 0, stream);
  launch_gemm(fc2o,  Bb, 1024, fcn_w, fcn_b, nullptr, nullptr, 30, nveco, 0, stream);
  finalize_kernel<<<1, 256, 0, stream>>>(transo, nveco, (float*)d_out, Bb * 10);
}
